// SelfAttention_65481071399422
// MI455X (gfx1250) — matrix operand order repaired and checked by reading, not compiled, not run on hardware
//
#include <hip/hip_runtime.h>
#include <hip/hip_bf16.h>
#include <cmath>
#include <cstdint>

typedef __bf16 bf16_t;
typedef __attribute__((ext_vector_type(16))) __bf16 v16bf;
typedef __attribute__((ext_vector_type(8)))  __bf16 v8bf;
typedef __attribute__((ext_vector_type(8)))  float  v8f;

union AFrag {           // 16x32 bf16 A-fragment (or 32x16 B-fragment) per lane
    v16bf v;
    v8bf  h[2];
    bf16_t e[16];
};
union F8 {              // 16x16 f32 C/D fragment per lane
    v8f   v;
    float f[8];
};

// ---------------------------------------------------------------------------
// CDNA5 async global->LDS copy (ASYNCcnt-tracked) + wait.
// ---------------------------------------------------------------------------
__device__ __forceinline__ void async_copy_b128(const void* gaddr, void* lds_ptr) {
    // Low 32 bits of a generic pointer to LDS are the LDS byte address.
    uint32_t lds_off = (uint32_t)(uintptr_t)lds_ptr;
    asm volatile("global_load_async_to_lds_b128 %0, %1, off"
                 :: "v"(lds_off), "v"(gaddr) : "memory");
}
__device__ __forceinline__ void wait_async0() {
    asm volatile("s_wait_asynccnt 0x0" ::: "memory");
}

// ---------------------------------------------------------------------------
// fp32 -> bf16 conversion, optionally also writing the transpose.
// ---------------------------------------------------------------------------
__global__ void cvt_bf16_kernel(const float* __restrict__ src,
                                bf16_t* __restrict__ drow,
                                bf16_t* __restrict__ dcol,
                                int rows, int cols) {
    size_t total = (size_t)rows * cols;
    for (size_t idx = (size_t)blockIdx.x * blockDim.x + threadIdx.x;
         idx < total; idx += (size_t)gridDim.x * blockDim.x) {
        int r = (int)(idx / cols);
        int c = (int)(idx % cols);
        bf16_t b = (bf16_t)src[idx];
        if (drow) drow[idx] = b;
        if (dcol) dcol[(size_t)c * rows + r] = b;
    }
}

// ---------------------------------------------------------------------------
// Projection GEMM: Out[N,D] = Xb[N,D] @ W, with Wt[D,D] = W^T.
// 8 waves/block; each wave owns a 32x64 tile (block tile 256x64).
// Shared 64x32 B-tile is async-DMA'd to LDS, double-buffered.
// grid = (N/256, D/64)
// ---------------------------------------------------------------------------
__global__ __launch_bounds__(256)
void proj_gemm_kernel(const bf16_t* __restrict__ Xb,
                      const bf16_t* __restrict__ Wt,
                      bf16_t* __restrict__ Out,
                      int N, int D) {
    const int lane = threadIdx.x & 31;
    const int wave = threadIdx.x >> 5;       // 0..7
    const int half = lane >> 4;
    const int l16  = lane & 15;

    const int m_base = blockIdx.x * 256 + wave * 32;
    const int n_base = blockIdx.y * 64;

    __shared__ __align__(16) bf16_t ldsB[2][64 * 32];   // [col][feat], 4 KB/buf

    // Each of the 256 threads async-copies one 16B chunk of the B tile.
    const int colL = threadIdx.x >> 2;   // 0..63
    const int seg  = threadIdx.x & 3;    // 0..3 (8 feats each)
    const bf16_t* gB = Wt + (size_t)(n_base + colL) * D + seg * 8;

    F8 acc[8];
#pragma unroll
    for (int t = 0; t < 8; ++t)
#pragma unroll
        for (int r = 0; r < 8; ++r) acc[t].f[r] = 0.0f;

    // prologue: stage tile for kb = 0
    async_copy_b128(gB, &ldsB[0][colL * 32 + seg * 8]);

    int buf = 0;
    for (int kb = 0; kb < D; kb += 32) {
        wait_async0();          // this wave's staged tile is in LDS
        __syncthreads();        // all waves' chunks visible; prev buf fully read
        if (kb + 32 < D)        // overlap next tile's DMA with this step's math
            async_copy_b128(gB + kb + 32, &ldsB[buf ^ 1][colL * 32 + seg * 8]);

        // Two A fragments (rows m_base..+15, m_base+16..+31) from global.
        const bf16_t* ar0 = Xb + (size_t)(m_base + l16)      * D + kb + half * 8;
        const bf16_t* ar1 = Xb + (size_t)(m_base + 16 + l16) * D + kb + half * 8;
        AFrag a0, a1;
        a0.h[0] = *(const v8bf*)(ar0); a0.h[1] = *(const v8bf*)(ar0 + 16);
        a1.h[0] = *(const v8bf*)(ar1); a1.h[1] = *(const v8bf*)(ar1 + 16);

#pragma unroll
        for (int nt = 0; nt < 4; ++nt) {
            const bf16_t* bp = &ldsB[buf][(nt * 16 + l16) * 32 + half * 8];
            AFrag b;
            b.h[0] = *(const v8bf*)(bp);     // ds_load_b128
            b.h[1] = *(const v8bf*)(bp + 16);
            acc[nt].v = __builtin_amdgcn_wmma_f32_16x16x32_bf16(
                false, a0.v, false, b.v, (short)0, acc[nt].v, false, false);
            acc[4 + nt].v = __builtin_amdgcn_wmma_f32_16x16x32_bf16(
                false, a1.v, false, b.v, (short)0, acc[4 + nt].v, false, false);
        }
        buf ^= 1;
    }

#pragma unroll
    for (int mt = 0; mt < 2; ++mt)
#pragma unroll
        for (int nt = 0; nt < 4; ++nt)
#pragma unroll
            for (int r = 0; r < 8; ++r) {
                int row = m_base + mt * 16 + r + 8 * half;
                int col = n_base + nt * 16 + l16;
                Out[(size_t)row * D + col] = (bf16_t)acc[mt * 4 + nt].f[r];
            }
}

// ---------------------------------------------------------------------------
// Flash attention: one block (8 waves) per 16 query rows.  Wave w owns
// D-slice [w*128, w*128+128) of the output.  Vt = X^T bf16 ([D][N]).
// grid = N/16, block = 256.
// ---------------------------------------------------------------------------
__global__ __launch_bounds__(256)
void flash_attn_kernel(const bf16_t* __restrict__ Q,
                       const bf16_t* __restrict__ K,
                       const bf16_t* __restrict__ Vt,
                       float* __restrict__ Out,
                       int N, int D, float scale) {
    const int lane = threadIdx.x & 31;
    const int wave = threadIdx.x >> 5;       // 0..7
    const int half = lane >> 4;
    const int l16  = lane & 15;
    const int qbase = blockIdx.x * 16;
    const int dsl   = wave * 128;            // this wave's D-slice

    __shared__ float Sred[8][16 * 32];       // per-wave partial S (16 KB)
    __shared__ float Sfull[16 * 32];         // reduced S, then exp(S-m) (2 KB)
    __shared__ float pred[16 * 16];          // partial max / partial sum (1 KB)
    __shared__ float rowm[16], rowl[16], rowcorr[16];

    const int srow = threadIdx.x >> 4;       // 0..15: row for stats phases
    const int scol = threadIdx.x & 15;       // 0..15: col pair (scol, scol+16)

    // Q fragments for this wave's 128-feature slice: constant over KV loop.
    AFrag aq[4];
#pragma unroll
    for (int ks = 0; ks < 4; ++ks) {
        const bf16_t* arow = Q + (size_t)(qbase + l16) * D + dsl + ks * 32 + half * 8;
        aq[ks].h[0] = *(const v8bf*)(arow);
        aq[ks].h[1] = *(const v8bf*)(arow + 16);
    }

    F8 acc[8];                               // O accumulator: 16 rows x 128 cols
#pragma unroll
    for (int c = 0; c < 8; ++c)
#pragma unroll
        for (int r = 0; r < 8; ++r) acc[c].f[r] = 0.0f;

    if (threadIdx.x < 16) { rowm[threadIdx.x] = -1e30f; rowl[threadIdx.x] = 0.0f; }
    __syncthreads();

    for (int kb = 0; kb < N; kb += 32) {
        // ---- partial S = Q_slice . K_slice^T  (two 16-key tiles) ----
        F8 s0, s1;
#pragma unroll
        for (int r = 0; r < 8; ++r) { s0.f[r] = 0.0f; s1.f[r] = 0.0f; }
#pragma unroll
        for (int ks = 0; ks < 4; ++ks) {
            const bf16_t* k0 = K + (size_t)(kb + l16)      * D + dsl + ks * 32 + half * 8;
            const bf16_t* k1 = K + (size_t)(kb + 16 + l16) * D + dsl + ks * 32 + half * 8;
            AFrag b0, b1;
            b0.h[0] = *(const v8bf*)(k0); b0.h[1] = *(const v8bf*)(k0 + 16);
            b1.h[0] = *(const v8bf*)(k1); b1.h[1] = *(const v8bf*)(k1 + 16);
            s0.v = __builtin_amdgcn_wmma_f32_16x16x32_bf16(
                false, aq[ks].v, false, b0.v, (short)0, s0.v, false, false);
            s1.v = __builtin_amdgcn_wmma_f32_16x16x32_bf16(
                false, aq[ks].v, false, b1.v, (short)0, s1.v, false, false);
        }
        if (kb + 32 < N)  // pull next K block toward the caches
            __builtin_prefetch(K + (size_t)(kb + 32 + l16) * D + dsl, 0, 1);

#pragma unroll
        for (int r = 0; r < 8; ++r) {
            int row = r + 8 * half;
            Sred[wave][row * 32 + l16]      = s0.f[r];
            Sred[wave][row * 32 + 16 + l16] = s1.f[r];
        }
        __syncthreads();                                   // b1

        // ---- reduce partials across the 8 waves ----
        for (int t = threadIdx.x; t < 512; t += 256) {
            float a = 0.0f;
#pragma unroll
            for (int w = 0; w < 8; ++w) a += Sred[w][t];
            Sfull[t] = a * scale;
        }
        __syncthreads();                                   // b2

        // ---- parallel row-max: 256 threads -> 16 partials per row ----
        pred[srow * 16 + scol] = fmaxf(Sfull[srow * 32 + scol],
                                       Sfull[srow * 32 + 16 + scol]);
        __syncthreads();                                   // b3

        if (threadIdx.x < 16) {
            int row = threadIdx.x;
            float bm = pred[row * 16];
            for (int j = 1; j < 16; ++j) bm = fmaxf(bm, pred[row * 16 + j]);
            float m_old = rowm[row];
            float m_new = fmaxf(m_old, bm);
            rowcorr[row] = __expf(m_old - m_new);
            rowm[row]    = m_new;
        }
        __syncthreads();                                   // b4

        // ---- exp pass (in place into Sfull) + O rescale ----
        {
            float mn = rowm[srow];
            float e0 = __expf(Sfull[srow * 32 + scol]      - mn);
            float e1 = __expf(Sfull[srow * 32 + 16 + scol] - mn);
            Sfull[srow * 32 + scol]      = e0;
            Sfull[srow * 32 + 16 + scol] = e1;
            pred[srow * 16 + scol] = e0 + e1;              // partial row sums
        }
#pragma unroll
        for (int r = 0; r < 8; ++r) {
            float f = rowcorr[r + 8 * half];
#pragma unroll
            for (int c = 0; c < 8; ++c) acc[c].f[r] *= f;
        }
        __syncthreads();                                   // b5

        if (threadIdx.x < 16) {
            int row = threadIdx.x;
            float s = pred[row * 16];
            for (int j = 1; j < 16; ++j) s += pred[row * 16 + j];
            rowl[row] = rowl[row] * rowcorr[row] + s;
        }

        // ---- build P fragment from the exp'd Sfull (no per-lane exp) ----
        AFrag p;
#pragma unroll
        for (int e = 0; e < 16; ++e) {
            int key = (e & 7) + ((e >> 3) << 4) + (half << 3);  // A K-pattern
            p.e[e] = (bf16_t)Sfull[l16 * 32 + key];
        }

        // ---- O_slice += P @ V_block_slice ----
#pragma unroll
        for (int c = 0; c < 8; ++c) {
            const bf16_t* vrow = Vt + (size_t)(dsl + c * 16 + l16) * N + kb + half * 8;
            AFrag bv;
            bv.h[0] = *(const v8bf*)(vrow);
            bv.h[1] = *(const v8bf*)(vrow + 16);
            acc[c].v = __builtin_amdgcn_wmma_f32_16x16x32_bf16(
                false, p.v, false, bv.v, (short)0, acc[c].v, false, false);
        }
    }
    __syncthreads();   // rowl final

    // ---- epilogue: divide by softmax denominator, store fp32 ----
#pragma unroll
    for (int c = 0; c < 8; ++c)
#pragma unroll
        for (int r = 0; r < 8; ++r) {
            int row = r + 8 * half;
            float v = acc[c].f[r] / rowl[row];
            Out[(size_t)(qbase + row) * D + dsl + c * 16 + l16] = v;
        }
}

// ---------------------------------------------------------------------------
extern "C" void kernel_launch(void* const* d_in, const int* in_sizes, int n_in,
                              void* d_out, int out_size, void* d_ws, size_t ws_size,
                              hipStream_t stream) {
    const float* X  = (const float*)d_in[0];
    const float* Wq = (const float*)d_in[1];
    const float* Wk = (const float*)d_in[2];
    float* Out = (float*)d_out;

    int D = 1;
    while ((long long)D * D < (long long)in_sizes[1]) D <<= 1;   // D = 1024
    int N = in_sizes[0] / D;                                     // N = 8192

    char* ws = (char*)d_ws;
    size_t szND = (size_t)N * D * sizeof(bf16_t);
    size_t szDD = (size_t)D * D * sizeof(bf16_t);
    bf16_t* Xb  = (bf16_t*)ws; ws += szND;   // X  bf16 row-major
    bf16_t* Xt  = (bf16_t*)ws; ws += szND;   // X^T bf16 (= V^T)
    bf16_t* Qb  = (bf16_t*)ws; ws += szND;   // Q  bf16 row-major
    bf16_t* Kb  = (bf16_t*)ws; ws += szND;   // K  bf16 row-major
    bf16_t* Wqt = (bf16_t*)ws; ws += szDD;   // Wq^T bf16
    bf16_t* Wkt = (bf16_t*)ws; ws += szDD;   // Wk^T bf16

    cvt_bf16_kernel<<<4096, 256, 0, stream>>>(X,  Xb,      Xt,  N, D);
    cvt_bf16_kernel<<<1024, 256, 0, stream>>>(Wq, nullptr, Wqt, D, D);
    cvt_bf16_kernel<<<1024, 256, 0, stream>>>(Wk, nullptr, Wkt, D, D);

    dim3 gproj(N / 256, D / 64);
    proj_gemm_kernel<<<gproj, 256, 0, stream>>>(Xb, Wqt, Qb, N, D);
    proj_gemm_kernel<<<gproj, 256, 0, stream>>>(Xb, Wkt, Kb, N, D);

    float scale = 1.0f / sqrtf((float)D);
    flash_attn_kernel<<<N / 16, 256, 0, stream>>>(Qb, Kb, Xt, Out, N, D, scale);
}
